// Attention_Mod_25116968747677
// MI455X (gfx1250) — matrix operand order repaired and checked by reading, not compiled, not run on hardware
//
#include <hip/hip_runtime.h>
#include <hip/hip_bf16.h>

// ---------------------------------------------------------------------------
// Fused SAGAN-style attention for MI455X (gfx1250, wave32, WMMA).
// B=4, C=512, H=W=64 -> N=4096, Cq=64.
//   proj_kernel: Q/K/V 1x1-conv projections, f32 -> bf16 workspace
//   attn_kernel: two-pass flash attention (stats pass + recompute/PV pass),
//                all matrix math on v_wmma_f32_16x16x32_bf16; K tiles staged
//                into LDS via GLOBAL_LOAD_ASYNC_TO_LDS_B128 (ASYNCcnt path).
// ---------------------------------------------------------------------------

typedef __bf16 bf16_t;
typedef __attribute__((ext_vector_type(16))) __bf16 v16bf;
typedef __attribute__((ext_vector_type(8)))  float  v8f;

#define BB 4
#define CC 512
#define NN 4096
#define CQQ 64

// -------- CDNA5 async global->LDS copy (ASYNCcnt-tracked, no VGPR data) ----
#if defined(__has_builtin)
#  if __has_builtin(__builtin_amdgcn_global_load_async_to_lds_b128)
#    define USE_ASYNC_LDS 1
#  endif
#endif

// builtin parameter type per hipcc diagnostic:
//   "__attribute__((__vector_size__(4 * sizeof(int)))) int __device__ *"
typedef int v4i_vs __attribute__((vector_size(16)));
typedef __attribute__((address_space(1))) v4i_vs* gas_v4i;   // global (AS1)
typedef __attribute__((address_space(3))) v4i_vs* las_v4i;   // LDS (AS3)

__device__ __forceinline__ void lds_copy16(const void* gsrc, void* ldst) {
#ifdef USE_ASYNC_LDS
    __builtin_amdgcn_global_load_async_to_lds_b128(
        (gas_v4i)gsrc, (las_v4i)ldst, /*imm offset*/0, /*cpol*/0);
#else
    *(uint4*)ldst = *(const uint4*)gsrc;
#endif
}

__device__ __forceinline__ void lds_copy_wait() {
#ifdef USE_ASYNC_LDS
#  if __has_builtin(__builtin_amdgcn_s_wait_asynccnt)
    __builtin_amdgcn_s_wait_asynccnt(0);
#  else
    asm volatile("s_wait_asynccnt 0x0" ::: "memory");
#  endif
#endif
}

union FragU {
    v16bf  v;
    bf16_t h[16];
    uint4  q[2];
};

// B-matrix fragment: 16 contiguous halves (lane holds col = lane&15,
// K = (lane>>4)*16 + i). 32 bytes = two 16B loads.
__device__ __forceinline__ v16bf frag_contig16(const bf16_t* p) {
    FragU u;
    u.q[0] = *(const uint4*)(p);
    u.q[1] = *(const uint4*)(p + 8);
    return u.v;
}

// A-matrix fragment: two 8-half runs (K = k0 + g*16 + j and k0 + g*16 + 8 + j).
__device__ __forceinline__ v16bf frag_A_runs(const bf16_t* p0, const bf16_t* p1) {
    FragU u;
    u.q[0] = *(const uint4*)(p0);
    u.q[1] = *(const uint4*)(p1);
    return u.v;
}

__device__ __forceinline__ v8f wmma_bf16(v16bf a, v16bf b, v8f c) {
    return __builtin_amdgcn_wmma_f32_16x16x32_bf16(
        /*neg_a=*/false, a, /*neg_b=*/false, b,
        /*c_mod=*/(short)0, c, /*reuse_a=*/false, /*reuse_b=*/false);
}

// ---------------------------------------------------------------------------
// Kernel 1: projections.  One 16x16 output tile per wave.
//   Q: out[n][cq] = sum_c Wq[cq][c] * x[c][n]   (stored transposed: A-layout)
//   K: out[m][cq] = sum_c Wk[cq][c] * x[c][m]   (stored transposed: B cols contiguous)
//   V: out[d][m]  = sum_c Wv[d][c]  * x[c][m]   (natural layout: B cols contiguous)
// Tile ids: [0,4096) Q, [4096,8192) K, [8192,40960) V.  Grid 5120 x 256.
// ---------------------------------------------------------------------------
__global__ __launch_bounds__(256) void proj_kernel(
    const float* __restrict__ x,  const float* __restrict__ Wq,
    const float* __restrict__ Wk, const float* __restrict__ Wv,
    bf16_t* __restrict__ qbf, bf16_t* __restrict__ kbf, bf16_t* __restrict__ vbf)
{
    const int lane = threadIdx.x & 31;
    const int wid  = threadIdx.x >> 5;
    const int l15  = lane & 15;
    const int g    = lane >> 4;
    const int tile = blockIdx.x * 8 + wid;

    const float* Wsel;
    bf16_t* dst;
    int b, mt, nt, transposed;
    if (tile < 4096) {
        b = tile >> 10; int rem = tile & 1023;
        mt = rem >> 8;  nt = rem & 255;
        Wsel = Wq; dst = qbf; transposed = 1;
    } else if (tile < 8192) {
        int t2 = tile - 4096;
        b = t2 >> 10; int rem = t2 & 1023;
        mt = rem >> 8; nt = rem & 255;
        Wsel = Wk; dst = kbf; transposed = 1;
    } else {
        int t2 = tile - 8192;
        b = t2 >> 13; int rem = t2 & 8191;
        mt = rem >> 8; nt = rem & 255;
        Wsel = Wv; dst = vbf; transposed = 0;
    }
    const int mbase = mt * 16, nbase = nt * 16;
    const float* wrow = Wsel + (size_t)(mbase + l15) * CC;       // A: row of W
    const float* xcol = x + (size_t)b * CC * NN + nbase + l15;   // B: col of x

    v8f acc = {};
    for (int c0 = 0; c0 < CC; c0 += 32) {
        FragU ua;
        #pragma unroll
        for (int j = 0; j < 8; ++j) ua.h[j]     = (bf16_t)wrow[c0 + g * 16 + j];
        #pragma unroll
        for (int j = 0; j < 8; ++j) ua.h[8 + j] = (bf16_t)wrow[c0 + g * 16 + 8 + j];
        FragU ub;
        #pragma unroll
        for (int j = 0; j < 16; ++j)
            ub.h[j] = (bf16_t)xcol[(size_t)(c0 + g * 16 + j) * NN];
        acc = wmma_bf16(ua.v, ub.v, acc);
    }

    if (transposed) {
        // dst[b][n][cq]: lane n = nbase+l15, cq = mbase + 8g + r  (contiguous 16B)
        bf16_t* p = dst + ((size_t)b * NN + nbase + l15) * CQQ + mbase + 8 * g;
        FragU o;
        #pragma unroll
        for (int r = 0; r < 8; ++r) o.h[r] = (bf16_t)acc[r];
        *(uint4*)p = o.q[0];
    } else {
        // dst[b][d][m]: lane m = nbase+l15, d = mbase + 8g + r
        #pragma unroll
        for (int r = 0; r < 8; ++r)
            dst[((size_t)b * CC + mbase + 8 * g + r) * NN + nbase + l15] = (bf16_t)acc[r];
    }
}

// ---------------------------------------------------------------------------
// Kernel 2: fused attention.  Grid (N/32, B) x 256 threads (8 waves).
// Block owns 32 queries x all 512 channels.  Wave (nsub, dgrp) owns a
// 16-query x 128-channel slab (8 x v8f accumulators).
// ---------------------------------------------------------------------------
__global__ __launch_bounds__(256) void attn_kernel(
    const float* __restrict__ x, const bf16_t* __restrict__ qbf,
    const bf16_t* __restrict__ kbf, const bf16_t* __restrict__ vbf,
    const float* __restrict__ gamma_p, float* __restrict__ out)
{
    __shared__ bf16_t Qs[32 * 64];       // 4 KB, loaded once
    __shared__ bf16_t Ks[32 * 64];       // 4 KB per key block
    __shared__ bf16_t Ps[8][16 * 32];    // 8 KB: per-wave P staging (D->A relayout)

    const int b    = blockIdx.y;
    const int n0   = blockIdx.x * 32;
    const int tid  = threadIdx.x;
    const int lane = tid & 31;
    const int wid  = tid >> 5;
    const int l15  = lane & 15;
    const int g    = lane >> 4;
    const int nsub = wid & 1;            // which 16-query subtile
    const int dgrp = wid >> 1;           // which 128-channel group

    const bf16_t* kbase = kbf + (size_t)b * NN * CQQ;
    const int idx = tid * 8;             // this thread's 8-half chunk of a tile

    {   // cooperative Q tile load: 32x64 halves, 16B per thread (async -> LDS)
        lds_copy16(qbf + ((size_t)b * NN + n0) * CQQ + idx, &Qs[idx]);
        lds_copy_wait();
    }
    __syncthreads();

    float rmax[8], rsum[8];
    #pragma unroll
    for (int r = 0; r < 8; ++r) { rmax[r] = -3.0e38f; rsum[r] = 0.0f; }

    const bf16_t* Qrow = &Qs[(nsub * 16 + l15) * 64];

    // ---------------- pass 1: softmax stats ----------------
    for (int m0 = 0; m0 < NN; m0 += 32) {
        __syncthreads();                 // previous block fully consumed
        lds_copy16(kbase + (size_t)m0 * CQQ + idx, &Ks[idx]);
        lds_copy_wait();
        __syncthreads();                 // all waves' chunks visible
        if (m0 + 32 < NN)
            __builtin_prefetch(kbase + (size_t)(m0 + 32) * CQQ + idx, 0, 1);

        v8f s0 = {}, s1 = {};
        #pragma unroll
        for (int c0 = 0; c0 < 64; c0 += 32) {     // K-dim = Cq = 64
            v16bf aQ = frag_A_runs(Qrow + c0 + g * 16, Qrow + c0 + g * 16 + 8);
            v16bf b0 = frag_contig16(&Ks[(l15)      * 64 + c0 + g * 16]);
            s0 = wmma_bf16(aQ, b0, s0);           // keys m0..m0+15
            v16bf b1 = frag_contig16(&Ks[(16 + l15) * 64 + c0 + g * 16]);
            s1 = wmma_bf16(aQ, b1, s1);           // keys m0+16..m0+31
        }
        #pragma unroll
        for (int r = 0; r < 8; ++r) {             // row = r + 8g, col = l15
            float t = fmaxf(s0[r], s1[r]);
            #pragma unroll
            for (int mask = 1; mask <= 8; mask <<= 1)
                t = fmaxf(t, __shfl_xor(t, mask));
            const float mn = fmaxf(rmax[r], t);
            float p = __expf(s0[r] - mn) + __expf(s1[r] - mn);
            #pragma unroll
            for (int mask = 1; mask <= 8; mask <<= 1)
                p += __shfl_xor(p, mask);
            rsum[r] = rsum[r] * __expf(rmax[r] - mn) + p;
            rmax[r] = mn;
        }
    }

    float rinv[8];
    #pragma unroll
    for (int r = 0; r < 8; ++r) rinv[r] = 1.0f / rsum[r];

    v8f zero = {};
    v8f oacc[8];
    #pragma unroll
    for (int dt = 0; dt < 8; ++dt) oacc[dt] = zero;

    bf16_t* Pw = Ps[wid];
    const bf16_t* vrow =
        vbf + ((size_t)b * CC + dgrp * 128 + l15) * NN;   // B cols of V^T

    // ---------------- pass 2: recompute + PV ----------------
    for (int m0 = 0; m0 < NN; m0 += 32) {
        __syncthreads();
        lds_copy16(kbase + (size_t)m0 * CQQ + idx, &Ks[idx]);
        lds_copy_wait();
        __syncthreads();

        v8f s0 = {}, s1 = {};
        #pragma unroll
        for (int c0 = 0; c0 < 64; c0 += 32) {
            v16bf aQ = frag_A_runs(Qrow + c0 + g * 16, Qrow + c0 + g * 16 + 8);
            v16bf b0 = frag_contig16(&Ks[(l15)      * 64 + c0 + g * 16]);
            s0 = wmma_bf16(aQ, b0, s0);
            v16bf b1 = frag_contig16(&Ks[(16 + l15) * 64 + c0 + g * 16]);
            s1 = wmma_bf16(aQ, b1, s1);
        }

        // normalize -> bf16 P tile; stage through per-wave LDS (same-wave
        // DS ops are in-order, no cross-wave barrier needed for Pw)
        #pragma unroll
        for (int r = 0; r < 8; ++r) {
            const float p0 = __expf(s0[r] - rmax[r]) * rinv[r];
            const float p1 = __expf(s1[r] - rmax[r]) * rinv[r];
            Pw[(r + 8 * g) * 32 + l15]      = (bf16_t)p0;
            Pw[(r + 8 * g) * 32 + 16 + l15] = (bf16_t)p1;
        }
        v16bf aP = frag_A_runs(&Pw[l15 * 32 + g * 16],
                               &Pw[l15 * 32 + g * 16 + 8]);

        const bf16_t* vb = vrow + m0 + g * 16;
        #pragma unroll
        for (int dt = 0; dt < 8; ++dt) {
            v16bf bV = frag_contig16(vb + (size_t)dt * 16 * NN);
            oacc[dt] = wmma_bf16(aP, bV, oacc[dt]);
        }
    }

    // ---------------- epilogue: y = gamma*out + x ----------------
    const float gamma = gamma_p[0];
    const int nb = n0 + nsub * 16 + 8 * g;       // + r gives query index
    #pragma unroll
    for (int dt = 0; dt < 8; ++dt) {
        const int c = dgrp * 128 + dt * 16 + l15;
        const float* xp = x   + ((size_t)b * CC + c) * NN + nb;
        float*       yp = out + ((size_t)b * CC + c) * NN + nb;
        #pragma unroll
        for (int r = 0; r < 8; ++r)
            yp[r] = gamma * oacc[dt][r] + xp[r];
    }
}

// ---------------------------------------------------------------------------
extern "C" void kernel_launch(void* const* d_in, const int* in_sizes, int n_in,
                              void* d_out, int out_size, void* d_ws, size_t ws_size,
                              hipStream_t stream) {
    (void)in_sizes; (void)n_in; (void)out_size; (void)ws_size;
    const float* x     = (const float*)d_in[0];
    const float* Wq    = (const float*)d_in[1];
    const float* Wk    = (const float*)d_in[2];
    const float* Wv    = (const float*)d_in[3];
    const float* gamma = (const float*)d_in[4];
    float* out = (float*)d_out;

    // workspace: Qbf (2MB) | Kbf (2MB) | Vbf (16MB) = 20MB bf16 staging
    bf16_t* qbf = (bf16_t*)d_ws;
    bf16_t* kbf = qbf + (size_t)BB * NN * CQQ;
    bf16_t* vbf = kbf + (size_t)BB * NN * CQQ;

    // 40960 wave-tiles total (Q 4096 + K 4096 + V 32768), 8 waves/block
    proj_kernel<<<5120, 256, 0, stream>>>(x, Wq, Wk, Wv, qbf, kbf, vbf);
    attn_kernel<<<dim3(NN / 32, BB), 256, 0, stream>>>(x, qbf, kbf, vbf, gamma, out);
}
